// Attention_43928925503623
// MI455X (gfx1250) — matrix operand order repaired and checked by reading, not compiled, not run on hardware
//
#include <hip/hip_runtime.h>
#include <hip/hip_bf16.h>

// ---------------------------------------------------------------------------
// Types matching CDNA5 WMMA / TDM builtin signatures (probe-confirmed)
// ---------------------------------------------------------------------------
typedef __attribute__((ext_vector_type(16))) __bf16 v16bf;
typedef __attribute__((ext_vector_type(8)))  __bf16 v8bf;
typedef __attribute__((ext_vector_type(8)))  float  v8f;
typedef __attribute__((ext_vector_type(4)))  unsigned int u32x4;
typedef __attribute__((ext_vector_type(4)))  int i32x4;
typedef __attribute__((ext_vector_type(8)))  int i32x8;

#define S_LEN     2048
#define DIM       4096
#define QKVO      6144   // Q_SIZE + 2*KV_SIZE
#define Q_SIZE    4096
#define KV_SIZE   1024
#define HEAD_DIM  128
#define N_HEADS   32

// Tensor Data Mover availability (device pass only; host pass parses fallback)
#if defined(__gfx1250__) && defined(__has_builtin)
#if __has_builtin(__builtin_amdgcn_tensor_load_to_lds) && \
    __has_builtin(__builtin_amdgcn_s_wait_tensorcnt)
#define HAVE_TDM 1
#endif
#endif

__device__ __forceinline__ v8f wmma_bf16(v16bf a, v16bf b, v8f c) {
    return __builtin_amdgcn_wmma_f32_16x16x32_bf16(
        false, a, false, b, (short)0, c, false, false);
}

// A-fragment (16x32 MxK) from row-major source. ISA 7.12.2 layout:
// lane<16: row=lane, elems 0..7=K(k0..k0+7), 8..15=K(k0+16..23); lane>=16 shifts K by 8.
__device__ __forceinline__ v16bf make_afrag(const __bf16* p, int stride,
                                            int row0, int k0, int lane) {
    int half = lane >> 4, m = lane & 15;
    const __bf16* base = p + (size_t)(row0 + m) * stride + k0 + 8 * half;
    v8bf lo = *(const v8bf*)(base);
    v8bf hi = *(const v8bf*)(base + 16);
    v16bf r;
#pragma unroll
    for (int i = 0; i < 8; ++i) { r[i] = lo[i]; r[i + 8] = hi[i]; }
    return r;
}

// B-fragment (32x16 KxN) from row-major [N][K] source.
// lane<16: col=lane, elems 0..7=K(k0..k0+7), 8..15=K(k0+16..23); lane>=16 shifts K by 8.
__device__ __forceinline__ v16bf make_bfrag(const __bf16* p, int stride,
                                            int n0, int k0, int lane) {
    int half = lane >> 4, m = lane & 15;
    const __bf16* base = p + (size_t)(n0 + m) * stride + k0 + 8 * half;
    v8bf lo = *(const v8bf*)(base);
    v8bf hi = *(const v8bf*)(base + 16);
    v16bf r;
#pragma unroll
    for (int i = 0; i < 8; ++i) { r[i] = lo[i]; r[i + 8] = hi[i]; }
    return r;
}

#ifdef HAVE_TDM
// Issue one TDM 2-D tile load, global -> LDS. All sizes in 8-byte units
// (D#.data_size = 3). Descriptor layout per ISA 08_async_tensor §8.3/8.4.
__device__ __forceinline__ void tdm_load_2d(unsigned lds_addr, const void* gptr,
                                            unsigned tile_w8, unsigned tile_h,
                                            unsigned tensor_w8, unsigned tensor_h,
                                            unsigned stride8) {
    unsigned long long ga = (unsigned long long)(uintptr_t)gptr;
    u32x4 g0;
    g0[0] = 1u;                                   // count=1 (valid), user mode
    g0[1] = lds_addr;                             // lds_addr [63:32]
    g0[2] = (unsigned)(ga & 0xFFFFFFFFu);         // global_addr lo
    g0[3] = (unsigned)((ga >> 32) & 0x1FFFFFFu)   // global_addr hi (57-bit)
            | (2u << 30);                         // type=2 ("image")
    // group1: data_size=3 (8B) @ [17:16]; tensor_dim0 @ [79:48];
    // tensor_dim1 @ [111:80]; tile_dim0 @ [127:112]; tile_dim1 @ [143:128];
    // tensor_dim0_stride @ [207:160]
    unsigned long long q0 = (3ull << 16) |
                            ((unsigned long long)(tensor_w8 & 0xFFFFu) << 48);
    unsigned long long q1 = ((unsigned long long)tensor_w8 >> 16) |
                            ((unsigned long long)tensor_h << 16) |
                            ((unsigned long long)(tile_w8 & 0xFFFFu) << 48);
    unsigned long long q2 = (unsigned long long)(tile_h & 0xFFFFu) |
                            ((unsigned long long)stride8 << 32);
    unsigned long long q3 = 0ull;
    i32x8 g1;
    g1[0] = (int)q0; g1[1] = (int)(q0 >> 32);
    g1[2] = (int)q1; g1[3] = (int)(q1 >> 32);
    g1[4] = (int)q2; g1[5] = (int)(q2 >> 32);
    g1[6] = (int)q3; g1[7] = (int)(q3 >> 32);
    i32x4 z4 = {};
#if __clang_major__ >= 23
    i32x8 z8 = {};
    __builtin_amdgcn_tensor_load_to_lds(g0, g1, z4, z4, z8, 0);
#else
    __builtin_amdgcn_tensor_load_to_lds(g0, g1, z4, z4, 0);
#endif
}
__device__ __forceinline__ unsigned lds_off(const void* p) {
    return (unsigned)(uintptr_t)p;  // generic LDS ptr: low 32 bits = LDS offset
}
#endif  // HAVE_TDM

// ---------------------------------------------------------------------------
// fp32 -> bf16 conversion
// ---------------------------------------------------------------------------
__global__ void cvt_bf16(const float* __restrict__ in, __bf16* __restrict__ out,
                         long long n) {
    long long i = (long long)blockIdx.x * blockDim.x + threadIdx.x;
    if (i < n) out[i] = (__bf16)in[i];
}

// ---------------------------------------------------------------------------
// Tiled bf16 GEMM: C[M,N] (f32) = A[M,K] x Bw[N,K]^T
// Block 128x128, 8 waves as 4(M) x 2(N); wave computes 32x64 (8 WMMA / k-step).
// Tiles staged to double-buffered LDS by the Tensor Data Mover (wave 0 issues),
// overlapped with WMMA on the other buffer.
// ---------------------------------------------------------------------------
#define GEMM_BM 128
#define GEMM_BN 128
#define GEMM_BK 32

__global__ __launch_bounds__(256) void gemm_bf16_f32(
    const __bf16* __restrict__ A, const __bf16* __restrict__ Bw,
    float* __restrict__ C, int M, int N, int K) {
    __shared__ __align__(16) __bf16 sA[2][GEMM_BM * GEMM_BK];  // 2 x 8 KB
    __shared__ __align__(16) __bf16 sB[2][GEMM_BN * GEMM_BK];  // 2 x 8 KB
    const int tid = threadIdx.x, lane = tid & 31, wave = tid >> 5;
    const int wm = wave >> 1, wn = wave & 1;     // 4 x 2 wave grid
    const int m0 = blockIdx.y * GEMM_BM;
    const int n0 = blockIdx.x * GEMM_BN;
    const __bf16* Abase = A + (size_t)m0 * K;
    const __bf16* Bbase = Bw + (size_t)n0 * K;

    v8f acc[2][4] = {};
    const int KT = K / GEMM_BK;

#ifdef HAVE_TDM
    const unsigned w8 = (unsigned)(K / 4);  // row width/stride in 8B units
    if (wave == 0) {
        tdm_load_2d(lds_off(&sA[0][0]), Abase, 8, GEMM_BM, w8, M, w8);
        tdm_load_2d(lds_off(&sB[0][0]), Bbase, 8, GEMM_BN, w8, N, w8);
        __builtin_amdgcn_s_wait_tensorcnt(0);
    }
    __syncthreads();
    for (int kt = 0; kt < KT; ++kt) {
        const int cur = kt & 1, nxt = cur ^ 1;
        if (wave == 0 && kt + 1 < KT) {
            tdm_load_2d(lds_off(&sA[nxt][0]), Abase + (kt + 1) * GEMM_BK,
                        8, GEMM_BM, w8, M, w8);
            tdm_load_2d(lds_off(&sB[nxt][0]), Bbase + (kt + 1) * GEMM_BK,
                        8, GEMM_BN, w8, N, w8);
        }
        v16bf af[2];
#pragma unroll
        for (int mt = 0; mt < 2; ++mt)
            af[mt] = make_afrag(sA[cur], GEMM_BK, wm * 32 + mt * 16, 0, lane);
#pragma unroll
        for (int nt = 0; nt < 4; ++nt) {
            v16bf bf = make_bfrag(sB[cur], GEMM_BK, wn * 64 + nt * 16, 0, lane);
#pragma unroll
            for (int mt = 0; mt < 2; ++mt)
                acc[mt][nt] = wmma_bf16(af[mt], bf, acc[mt][nt]);
        }
        if (wave == 0 && kt + 1 < KT) __builtin_amdgcn_s_wait_tensorcnt(0);
        __syncthreads();
    }
#else
    for (int kt = 0; kt < KT; ++kt) {
        const int k0 = kt * GEMM_BK;
        __syncthreads();
#pragma unroll
        for (int ch = tid; ch < (GEMM_BM * GEMM_BK / 8); ch += 256) {
            int r = ch >> 2, c = (ch & 3) * 8;
            *(v8bf*)&sA[0][r * GEMM_BK + c] =
                *(const v8bf*)&A[(size_t)(m0 + r) * K + k0 + c];
        }
#pragma unroll
        for (int ch = tid; ch < (GEMM_BN * GEMM_BK / 8); ch += 256) {
            int r = ch >> 2, c = (ch & 3) * 8;
            *(v8bf*)&sB[0][r * GEMM_BK + c] =
                *(const v8bf*)&Bw[(size_t)(n0 + r) * K + k0 + c];
        }
        __syncthreads();
        v16bf af[2];
#pragma unroll
        for (int mt = 0; mt < 2; ++mt)
            af[mt] = make_afrag(sA[0], GEMM_BK, wm * 32 + mt * 16, 0, lane);
#pragma unroll
        for (int nt = 0; nt < 4; ++nt) {
            v16bf bf = make_bfrag(sB[0], GEMM_BK, wn * 64 + nt * 16, 0, lane);
#pragma unroll
            for (int mt = 0; mt < 2; ++mt)
                acc[mt][nt] = wmma_bf16(af[mt], bf, acc[mt][nt]);
        }
    }
#endif

    const int half = lane >> 4, mcol = lane & 15;
#pragma unroll
    for (int mt = 0; mt < 2; ++mt)
#pragma unroll
        for (int nt = 0; nt < 4; ++nt)
#pragma unroll
            for (int r = 0; r < 8; ++r)
                C[(size_t)(m0 + wm * 32 + mt * 16 + r + 8 * half) * N +
                  n0 + wn * 64 + nt * 16 + mcol] = acc[mt][nt][r];
}

// ---------------------------------------------------------------------------
// RoPE on Q and K -> bf16 Q [s][4096], K [kvh][s][128]
// ---------------------------------------------------------------------------
__global__ void rope_qk(const float* __restrict__ qkv,
                        const float* __restrict__ fcos,
                        const float* __restrict__ fsin,
                        __bf16* __restrict__ Qb, __bf16* __restrict__ Kb) {
    const int PAIRS = Q_SIZE / 2 + KV_SIZE / 2;
    long long idx = (long long)blockIdx.x * blockDim.x + threadIdx.x;
    if (idx >= (long long)S_LEN * PAIRS) return;
    int s = (int)(idx / PAIRS), p = (int)(idx % PAIRS);
    if (p < Q_SIZE / 2) {
        int head = p >> 6, i = p & 63;
        int d0 = head * HEAD_DIM + 2 * i;
        float a = qkv[(size_t)s * QKVO + d0];
        float b = qkv[(size_t)s * QKVO + d0 + 1];
        float c = fcos[s * 64 + i], sn = fsin[s * 64 + i];
        Qb[(size_t)s * Q_SIZE + d0]     = (__bf16)(a * c - b * sn);
        Qb[(size_t)s * Q_SIZE + d0 + 1] = (__bf16)(a * sn + b * c);
    } else {
        int kp = p - Q_SIZE / 2;
        int kvh = kp >> 6, i = kp & 63;
        int srcd = Q_SIZE + kvh * HEAD_DIM + 2 * i;
        float a = qkv[(size_t)s * QKVO + srcd];
        float b = qkv[(size_t)s * QKVO + srcd + 1];
        float c = fcos[s * 64 + i], sn = fsin[s * 64 + i];
        __bf16* kb = Kb + ((size_t)kvh * S_LEN + s) * HEAD_DIM + 2 * i;
        kb[0] = (__bf16)(a * c - b * sn);
        kb[1] = (__bf16)(a * sn + b * c);
    }
}

__global__ void v_pack(const float* __restrict__ qkv, __bf16* __restrict__ Vb) {
    long long idx = (long long)blockIdx.x * blockDim.x + threadIdx.x;
    if (idx >= (long long)S_LEN * KV_SIZE) return;
    int s = (int)(idx / KV_SIZE), e = (int)(idx % KV_SIZE);
    int kvh = e >> 7, d = e & 127;
    Vb[((size_t)kvh * S_LEN + s) * HEAD_DIM + d] =
        (__bf16)qkv[(size_t)s * QKVO + Q_SIZE + KV_SIZE + e];
}

// ---------------------------------------------------------------------------
// Flash attention (causal), bf16 WMMA, online softmax. K tile via TDM.
// Grid: (S/128, N_HEADS); block = 256 (8 waves); wave -> 16 query rows.
// ---------------------------------------------------------------------------
#define FA_BQ 128
#define FA_BK 32

__global__ __launch_bounds__(256) void flash_fwd(
    const __bf16* __restrict__ Q, const __bf16* __restrict__ Kc,
    const __bf16* __restrict__ Vc, __bf16* __restrict__ O) {
    __shared__ __align__(16) __bf16 sK[FA_BK * HEAD_DIM];    // [t][d]  8 KB
    __shared__ __align__(16) __bf16 sVt[HEAD_DIM * FA_BK];   // [d][t]  8 KB
    __shared__ __align__(16) __bf16 sP[8 * 16 * FA_BK];      // per-wave P 8 KB

    const int tid = threadIdx.x, lane = tid & 31, wave = tid >> 5;
    const int half = lane >> 4, mcol = lane & 15;
    const int h = blockIdx.y;
    const int kvh = h >> 2;  // N_REP = 4
    const int qblk = blockIdx.x;
    const int qbase = qblk * FA_BQ + wave * 16;
    const float scale = 0.08838834764831845f;  // 1/sqrt(128)
    const __bf16* Kh = Kc + (size_t)kvh * S_LEN * HEAD_DIM;
    const __bf16* Vh = Vc + (size_t)kvh * S_LEN * HEAD_DIM;

    v16bf qf[4];
    const __bf16* qptr = Q + (size_t)qbase * Q_SIZE + h * HEAD_DIM;
#pragma unroll
    for (int ds = 0; ds < 4; ++ds) qf[ds] = make_afrag(qptr, Q_SIZE, 0, ds * 32, lane);

    v8f o[8] = {};
    float mrow[8], lrow[8];
#pragma unroll
    for (int r = 0; r < 8; ++r) { mrow[r] = -3.0e38f; lrow[r] = 0.f; }

    const int nTiles = (qblk + 1) * (FA_BQ / FA_BK);

    for (int kt = 0; kt < nTiles; ++kt) {
        const int t0 = kt * FA_BK;
        __syncthreads();
        // K tile 32x128 (row-contiguous): TDM DMA if available
#ifdef HAVE_TDM
        if (wave == 0)
            tdm_load_2d(lds_off(&sK[0]), Kh + (size_t)t0 * HEAD_DIM,
                        32, FA_BK, 32, S_LEN, 32);
#else
#pragma unroll
        for (int ch = tid; ch < 512; ch += 256) {
            int r = ch >> 4, c = (ch & 15) * 8;
            *(v8bf*)&sK[r * HEAD_DIM + c] =
                *(const v8bf*)&Kh[(size_t)(t0 + r) * HEAD_DIM + c];
        }
#endif
        // V tile, transposed into LDS (TDM cannot transpose)
#pragma unroll
        for (int ch = tid; ch < 512; ch += 256) {
            int r = ch >> 4, c = (ch & 15) * 8;
            v8bf v = *(const v8bf*)&Vh[(size_t)(t0 + r) * HEAD_DIM + c];
#pragma unroll
            for (int j = 0; j < 8; ++j) sVt[(c + j) * FA_BK + r] = v[j];
        }
#ifdef HAVE_TDM
        if (wave == 0) __builtin_amdgcn_s_wait_tensorcnt(0);
#endif
        __syncthreads();

        // S = Q K^T : 2 N-tiles x 4 k-steps -> 8 WMMA
        v8f sc[2] = {};
#pragma unroll
        for (int ds = 0; ds < 4; ++ds) {
#pragma unroll
            for (int nt = 0; nt < 2; ++nt) {
                v16bf bf = make_bfrag(sK, HEAD_DIM, nt * 16, ds * 32, lane);
                sc[nt] = wmma_bf16(qf[ds], bf, sc[nt]);
            }
        }

        // Online softmax (rows live per lane-half; 16-lane shfl_xor reductions)
        float corr[8];
#pragma unroll
        for (int r = 0; r < 8; ++r) {
            int q = qbase + r + 8 * half;
            float s0 = sc[0][r] * scale + ((t0 + mcol) > q ? -1e9f : 0.f);
            float s1 = sc[1][r] * scale + ((t0 + 16 + mcol) > q ? -1e9f : 0.f);
            float tmax = fmaxf(s0, s1);
#pragma unroll
            for (int off = 1; off < 16; off <<= 1)
                tmax = fmaxf(tmax, __shfl_xor(tmax, off, 32));
            float mnew = fmaxf(mrow[r], tmax);
            float c = __expf(mrow[r] - mnew);
            float p0 = __expf(s0 - mnew), p1 = __expf(s1 - mnew);
            float rs = p0 + p1;
#pragma unroll
            for (int off = 1; off < 16; off <<= 1)
                rs += __shfl_xor(rs, off, 32);
            lrow[r] = lrow[r] * c + rs;
            mrow[r] = mnew;
            corr[r] = c;
            __bf16* pb = &sP[(wave * 16 + r + 8 * half) * FA_BK];
            pb[mcol]      = (__bf16)p0;
            pb[16 + mcol] = (__bf16)p1;
        }
        __syncthreads();

        v16bf pf = make_afrag(&sP[wave * 16 * FA_BK], FA_BK, 0, 0, lane);
#pragma unroll
        for (int dt = 0; dt < 8; ++dt) {
            v16bf bf = make_bfrag(sVt, FA_BK, dt * 16, 0, lane);
            v8f on = o[dt];
#pragma unroll
            for (int r = 0; r < 8; ++r) on[r] *= corr[r];
            o[dt] = wmma_bf16(pf, bf, on);
        }
    }

#pragma unroll
    for (int dt = 0; dt < 8; ++dt)
#pragma unroll
        for (int r = 0; r < 8; ++r) {
            float val = o[dt][r] / lrow[r];
            O[(size_t)(qbase + r + 8 * half) * Q_SIZE + h * HEAD_DIM + dt * 16 + mcol] =
                (__bf16)val;
        }
}

// ---------------------------------------------------------------------------
// Launcher
// ---------------------------------------------------------------------------
extern "C" void kernel_launch(void* const* d_in, const int* in_sizes, int n_in,
                              void* d_out, int out_size, void* d_ws, size_t ws_size,
                              hipStream_t stream) {
    const float* x    = (const float*)d_in[0];
    const float* fcos = (const float*)d_in[1];
    const float* fsin = (const float*)d_in[2];
    // d_in[3] = mask: causal mask applied analytically
    const float* wqkv = (const float*)d_in[4];
    const float* wo   = (const float*)d_in[5];
    float* out = (float*)d_out;

    char* ws = (char*)d_ws;
    size_t off = 0;
    auto carve = [&](size_t bytes) -> void* {
        void* p = ws + off;
        off += (bytes + 255) & ~(size_t)255;
        return p;
    };
    float*  qkv_f  = (float*)carve((size_t)S_LEN * QKVO * 4);
    __bf16* wqkv_b = (__bf16*)carve((size_t)QKVO * DIM * 2);
    __bf16* wo_b   = (__bf16*)carve((size_t)DIM * DIM * 2);
    __bf16* xb     = (__bf16*)carve((size_t)S_LEN * DIM * 2);
    __bf16* qb     = (__bf16*)carve((size_t)S_LEN * Q_SIZE * 2);
    __bf16* kb     = (__bf16*)carve((size_t)S_LEN * KV_SIZE * 2);
    __bf16* vb     = (__bf16*)carve((size_t)S_LEN * KV_SIZE * 2);
    __bf16* attnb  = xb;  // xb dead after QKV GEMM -> reuse

    auto blocks = [](long long n) { return (unsigned)((n + 255) / 256); };

    cvt_bf16<<<blocks((long long)S_LEN * DIM), 256, 0, stream>>>(x, xb, (long long)S_LEN * DIM);
    cvt_bf16<<<blocks((long long)QKVO * DIM), 256, 0, stream>>>(wqkv, wqkv_b, (long long)QKVO * DIM);
    cvt_bf16<<<blocks((long long)DIM * DIM), 256, 0, stream>>>(wo, wo_b, (long long)DIM * DIM);

    gemm_bf16_f32<<<dim3(QKVO / GEMM_BN, S_LEN / GEMM_BM), 256, 0, stream>>>(
        xb, wqkv_b, qkv_f, S_LEN, QKVO, DIM);

    rope_qk<<<blocks((long long)S_LEN * (Q_SIZE / 2 + KV_SIZE / 2)), 256, 0, stream>>>(
        qkv_f, fcos, fsin, qb, kb);
    v_pack<<<blocks((long long)S_LEN * KV_SIZE), 256, 0, stream>>>(qkv_f, vb);

    flash_fwd<<<dim3(S_LEN / FA_BQ, N_HEADS), 256, 0, stream>>>(qb, kb, vb, attnb);

    gemm_bf16_f32<<<dim3(DIM / GEMM_BN, S_LEN / GEMM_BM), 256, 0, stream>>>(
        attnb, wo_b, out, S_LEN, DIM, DIM);
}
